// E_GCL_49976239456638
// MI455X (gfx1250) — matrix operand order repaired and checked by reading, not compiled, not run on hardware
//
#include <hip/hip_runtime.h>

typedef __attribute__((ext_vector_type(16))) __bf16 v16bf;
typedef __attribute__((ext_vector_type(8)))  float  v8f;

#define WMMA_BF16(a,b,c) __builtin_amdgcn_wmma_f32_16x16x32_bf16(false,(a),false,(b),(short)0,(c),false,false)

#define Nn 50000
#define Ee 800000
#define STR 136   // LDS row stride (elements); 272B rows -> 16B-aligned fragments

// packed-weight element offsets (bf16 elements) inside workspace
#define MW0P 0        // 256x128 (radial row handled in f32 epilogue), Kt=8
#define MW1P 32768    // 128x128, Kt=4
#define CW0P 49152
#define CW1P 65536
#define NW0P 81920    // 256x128, Kt=8
#define NW1P 114688
#define NW2P 131072
#define WPACK_BYTES 294912            // 147456 bf16 elements
#define AGG_OFF  ((size_t)WPACK_BYTES)            // N*128 f32
#define TS_OFF   (AGG_OFF + (size_t)Nn*128*4)     // N*3 f32
#define CNT_OFF  (TS_OFF  + (size_t)Nn*3*4)       // N f32

__device__ __forceinline__ v8f splat8(float x){
  v8f r;
  #pragma unroll
  for (int i=0;i<8;++i) r[i]=x;
  return r;
}

__device__ __forceinline__ unsigned pkbf(float lo, float hi){
  union { __bf16 b[2]; unsigned u; } t;
  t.b[0]=(__bf16)lo; t.b[1]=(__bf16)hi;
  return t.u;
}

__device__ __forceinline__ void atomAddF(float* p, float v){
  __hip_atomic_fetch_add(p, v, __ATOMIC_RELAXED, __HIP_MEMORY_SCOPE_AGENT);
}

// Build 16x32 bf16 A-fragment from a per-lane global f32 row pointer.
// K pattern per half-wave g: [kloc+8g .. +7] and [kloc+16+8g .. +7] -> 4x float4.
__device__ __forceinline__ v16bf gfrag(const float* __restrict__ rowp, int kloc, int lane){
  union { unsigned u[8]; v16bf v; } U;
  int g8 = (lane >> 4) * 8;
  float4 a0 = *(const float4*)(rowp + kloc + g8);
  float4 a1 = *(const float4*)(rowp + kloc + g8 + 4);
  float4 b0 = *(const float4*)(rowp + kloc + 16 + g8);
  float4 b1 = *(const float4*)(rowp + kloc + 16 + g8 + 4);
  U.u[0]=pkbf(a0.x,a0.y); U.u[1]=pkbf(a0.z,a0.w);
  U.u[2]=pkbf(a1.x,a1.y); U.u[3]=pkbf(a1.z,a1.w);
  U.u[4]=pkbf(b0.x,b0.y); U.u[5]=pkbf(b0.z,b0.w);
  U.u[6]=pkbf(b1.x,b1.y); U.u[7]=pkbf(b1.z,b1.w);
  return U.v;
}

__device__ __forceinline__ v16bf gfrag_scaled(const float* __restrict__ rowp, int kloc, int lane, float s){
  union { unsigned u[8]; v16bf v; } U;
  int g8 = (lane >> 4) * 8;
  float4 a0 = *(const float4*)(rowp + kloc + g8);
  float4 a1 = *(const float4*)(rowp + kloc + g8 + 4);
  float4 b0 = *(const float4*)(rowp + kloc + 16 + g8);
  float4 b1 = *(const float4*)(rowp + kloc + 16 + g8 + 4);
  U.u[0]=pkbf(a0.x*s,a0.y*s); U.u[1]=pkbf(a0.z*s,a0.w*s);
  U.u[2]=pkbf(a1.x*s,a1.y*s); U.u[3]=pkbf(a1.z*s,a1.w*s);
  U.u[4]=pkbf(b0.x*s,b0.y*s); U.u[5]=pkbf(b0.z*s,b0.w*s);
  U.u[6]=pkbf(b1.x*s,b1.y*s); U.u[7]=pkbf(b1.z*s,b1.w*s);
  return U.v;
}

// A-fragment from LDS bf16 16xSTR tile: two 16B ds loads.
__device__ __forceinline__ v16bf lds_afrag(const __bf16* A, int lane, int kT){
  union { uint4 q[2]; v16bf v; } U;
  int m = lane & 15, g8 = (lane >> 4) * 8;
  const __bf16* p = A + m*STR + kT*32 + g8;
  U.q[0] = *(const uint4*)(p);        // K = base + 0..7
  U.q[1] = *(const uint4*)(p + 16);   // K = base + 16..23
  return U.v;
}

// C layout: lane holds (M = r + 8*(l/16), N16 = l%16) for r=0..7, per N-tile t.
__device__ __forceinline__ void store_silu(__bf16* O, const v8f acc[8], int lane){
  int n = lane & 15, g = lane >> 4;
  #pragma unroll
  for (int t=0;t<8;++t){
    #pragma unroll
    for (int r=0;r<8;++r){
      float x = acc[t][r];
      x = x * (1.0f/(1.0f + __expf(-x)));
      O[(r + 8*g)*STR + t*16 + n] = (__bf16)x;
    }
  }
}

// One 16x128x128 GEMM stage: A from LDS, packed bf16 weights, bias init.
__device__ __forceinline__ void gemm_lds_stage(const __bf16* A, const __bf16* __restrict__ Wt,
                                               const float* __restrict__ bias, int lane, v8f acc[8]){
  int n = lane & 15;
  #pragma unroll
  for (int t=0;t<8;++t) acc[t] = splat8(bias[t*16+n]);
  for (int kT=0;kT<4;++kT){
    v16bf a = lds_afrag(A, lane, kT);
    #pragma unroll
    for (int t=0;t<8;++t){
      v16bf b = *(const v16bf*)(Wt + (((size_t)(t*4+kT))<<9) + (lane<<4));
      acc[t] = WMMA_BF16(a, b, acc[t]);
    }
  }
}

// ---------------- weight packing: f32 row-major (K x 128) -> bf16 B-fragments ----------------
__global__ void egcl_pack(const float* __restrict__ src, __bf16* __restrict__ dst, int Kact, int Kt){
  int t = blockIdx.x*blockDim.x + threadIdx.x;
  int total = Kt*4096;
  if (t >= total) return;
  int frag = t >> 9, rem = t & 511;
  int lane = rem >> 4, j = rem & 15;
  int nT = frag / Kt, kT = frag % Kt;
  int k = kT*32 + (lane>>4)*8 + (j & 7) + ((j >> 3) << 4);
  int nn = nT*16 + (lane&15);
  float v = (k < Kact) ? src[(size_t)k*128 + nn] : 0.0f;
  dst[t] = (__bf16)v;
}

// ---------------- edge kernel: 4 waves x 16 edges ----------------
__global__ void __launch_bounds__(128) egcl_edge(
    const float* __restrict__ h, const float* __restrict__ coord,
    const long long* __restrict__ ei, const __bf16* __restrict__ wp,
    const float* __restrict__ mw0raw,
    const float* __restrict__ mb0, const float* __restrict__ mb1,
    const float* __restrict__ cb0, const float* __restrict__ cb1,
    const float* __restrict__ cw2,
    float* __restrict__ agg, float* __restrict__ tsum, float* __restrict__ cnt)
{
  __shared__ alignas(16) __bf16 sA[4][16*STR];
  __shared__ alignas(16) __bf16 sB[4][16*STR];   // holds final m
  __shared__ int   sRow[4][16], sCol[4][16];
  __shared__ float sRad[4][16], sCd[4][48], sPhi[4][16];

  int tid = threadIdx.x, wave = tid >> 5, lane = tid & 31;
  int n = lane & 15, m = lane & 15, g = lane >> 4;
  long long e0 = (long long)(blockIdx.x*4 + wave)*16;

  if (lane < 16){
    long long e = e0 + lane;
    int r = (int)ei[2*e], c = (int)ei[2*e+1];
    sRow[wave][lane] = r; sCol[wave][lane] = c;
    float dx = coord[r*3+0]-coord[c*3+0];
    float dy = coord[r*3+1]-coord[c*3+1];
    float dz = coord[r*3+2]-coord[c*3+2];
    sCd[wave][lane*3+0]=dx; sCd[wave][lane*3+1]=dy; sCd[wave][lane*3+2]=dz;
    sRad[wave][lane] = dx*dx + dy*dy + dz*dz;
  }
  __syncthreads();

  // stage 1: [h[row] | h[col]] (16x256) @ mw0p + mb0, radial rank-1 in f32, silu
  v8f acc[8];
  #pragma unroll
  for (int t=0;t<8;++t) acc[t] = splat8(mb0[t*16+n]);
  for (int kT=0;kT<8;++kT){
    int node = (kT<4 ? sRow : sCol)[wave][m];
    v16bf a = gfrag(h + (size_t)node*128, (kT&3)*32, lane);
    #pragma unroll
    for (int t=0;t<8;++t){
      v16bf b = *(const v16bf*)(wp + MW0P + (((size_t)(t*8+kT))<<9) + (lane<<4));
      acc[t] = WMMA_BF16(a, b, acc[t]);
    }
  }
  {
    // radial (K=256) contribution: acc[t][r] += radial[M] * mw0[256][t*16+n]  (full f32)
    float radv[8];
    #pragma unroll
    for (int r=0;r<8;++r) radv[r] = sRad[wave][r + 8*g];
    const float* w256 = mw0raw + (size_t)256*128;
    #pragma unroll
    for (int t=0;t<8;++t){
      float w = w256[t*16+n];
      #pragma unroll
      for (int r=0;r<8;++r) acc[t][r] += radv[r]*w;
    }
  }
  store_silu(sA[wave], acc, lane);
  __syncthreads();

  // stage 2: m = silu(. @ mw1 + mb1) -> sB (kept for agg)
  gemm_lds_stage(sA[wave], wp+MW1P, mb1, lane, acc);
  store_silu(sB[wave], acc, lane);
  __syncthreads();

  // stage 3: silu(m @ cw0 + cb0) -> sA
  gemm_lds_stage(sB[wave], wp+CW0P, cb0, lane, acc);
  store_silu(sA[wave], acc, lane);
  __syncthreads();

  // stage 4: silu(. @ cw1 + cb1), fused phi_x = . @ cw2 in epilogue
  gemm_lds_stage(sA[wave], wp+CW1P, cb1, lane, acc);
  float phi[8];
  #pragma unroll
  for (int r=0;r<8;++r) phi[r]=0.0f;
  #pragma unroll
  for (int t=0;t<8;++t){
    float w = cw2[t*16+n];
    #pragma unroll
    for (int r=0;r<8;++r){
      float x = acc[t][r];
      x = x * (1.0f/(1.0f + __expf(-x)));
      phi[r] += x * w;
    }
  }
  #pragma unroll
  for (int mk=1; mk<16; mk<<=1){
    #pragma unroll
    for (int r=0;r<8;++r) phi[r] += __shfl_xor(phi[r], mk, 32);
  }
  if (n == 0){   // lanes 0 and 16 hold M=0..7 / 8..15 sums
    #pragma unroll
    for (int r=0;r<8;++r) sPhi[wave][g*8+r] = phi[r];
  }
  __syncthreads();

  // scatter: trans into tsum/cnt, m into agg
  if (lane < 16){
    int r = sRow[wave][lane];
    float p = sPhi[wave][lane];
    atomAddF(&tsum[(size_t)r*3+0], sCd[wave][lane*3+0]*p);
    atomAddF(&tsum[(size_t)r*3+1], sCd[wave][lane*3+1]*p);
    atomAddF(&tsum[(size_t)r*3+2], sCd[wave][lane*3+2]*p);
    atomAddF(&cnt[r], 1.0f);
  }
  #pragma unroll 1
  for (int e=0;e<16;++e){
    int r = sRow[wave][e];
    float* dst = agg + (size_t)r*128 + lane*4;
    const __bf16* src = &sB[wave][e*STR + lane*4];
    #pragma unroll
    for (int f=0;f<4;++f) atomAddF(dst+f, (float)src[f]);
  }
}

// ---------------- node kernel: 4 waves x 16 nodes ----------------
__global__ void __launch_bounds__(128) egcl_node(
    const float* __restrict__ h, const __bf16* __restrict__ wp,
    const float* __restrict__ nb0, const float* __restrict__ nb1, const float* __restrict__ nb2,
    const float* __restrict__ agg, const float* __restrict__ cnt, float* __restrict__ out)
{
  __shared__ alignas(16) __bf16 sA[4][16*STR];
  __shared__ alignas(16) __bf16 sB[4][16*STR];
  int tid = threadIdx.x, wave = tid >> 5, lane = tid & 31;
  int n = lane & 15, m = lane & 15, g = lane >> 4;
  int n0 = (blockIdx.x*4 + wave)*16;
  int node_m = n0 + m; if (node_m >= Nn) node_m = Nn-1;
  float inv = 1.0f / fmaxf(cnt[node_m], 1.0f);

  // stage 1: [h | agg/cnt] (16x256) @ nw0 (256x128) + nb0, silu
  v8f acc[8];
  #pragma unroll
  for (int t=0;t<8;++t) acc[t] = splat8(nb0[t*16+n]);
  for (int kT=0;kT<8;++kT){
    v16bf a = (kT<4) ? gfrag(h + (size_t)node_m*128, kT*32, lane)
                     : gfrag_scaled(agg + (size_t)node_m*128, (kT-4)*32, lane, inv);
    #pragma unroll
    for (int t=0;t<8;++t){
      v16bf b = *(const v16bf*)(wp + NW0P + (((size_t)(t*8+kT))<<9) + (lane<<4));
      acc[t] = WMMA_BF16(a, b, acc[t]);
    }
  }
  store_silu(sA[wave], acc, lane);
  __syncthreads();

  gemm_lds_stage(sA[wave], wp+NW1P, nb1, lane, acc);
  store_silu(sB[wave], acc, lane);
  __syncthreads();

  // final stage: no silu; h_new = h + (. @ nw2 + nb2)
  gemm_lds_stage(sB[wave], wp+NW2P, nb2, lane, acc);
  #pragma unroll
  for (int t=0;t<8;++t){
    #pragma unroll
    for (int r=0;r<8;++r){
      int node = n0 + r + 8*g;
      if (node < Nn){
        int col = t*16 + n;
        out[(size_t)node*128 + col] = h[(size_t)node*128 + col] + acc[t][r];
      }
    }
  }
}

// ---------------- coord update ----------------
__global__ void egcl_coord(const float* __restrict__ coord, const float* __restrict__ tsum,
                           const float* __restrict__ cnt, float* __restrict__ outc)
{
  int i = blockIdx.x*blockDim.x + threadIdx.x;
  if (i < Nn*3){
    int nd = i/3;
    outc[i] = coord[i] + tsum[i] / fmaxf(cnt[nd], 1.0f);
  }
}

extern "C" void kernel_launch(void* const* d_in, const int* in_sizes, int n_in,
                              void* d_out, int out_size, void* d_ws, size_t ws_size,
                              hipStream_t stream)
{
  const float* h   = (const float*)d_in[0];
  const float* crd = (const float*)d_in[1];
  const float* mw0 = (const float*)d_in[2];
  const float* mb0 = (const float*)d_in[3];
  const float* mw1 = (const float*)d_in[4];
  const float* mb1 = (const float*)d_in[5];
  const float* nw0 = (const float*)d_in[6];
  const float* nb0 = (const float*)d_in[7];
  const float* nw1 = (const float*)d_in[8];
  const float* nb1 = (const float*)d_in[9];
  const float* nw2 = (const float*)d_in[10];
  const float* nb2 = (const float*)d_in[11];
  const float* cw0 = (const float*)d_in[12];
  const float* cb0 = (const float*)d_in[13];
  const float* cw1 = (const float*)d_in[14];
  const float* cb1 = (const float*)d_in[15];
  const float* cw2 = (const float*)d_in[16];
  const long long* ei = (const long long*)d_in[17];
  float* out = (float*)d_out;
  char* ws = (char*)d_ws;

  __bf16* wp  = (__bf16*)ws;
  float* agg  = (float*)(ws + AGG_OFF);
  float* tsum = (float*)(ws + TS_OFF);
  float* cnt  = (float*)(ws + CNT_OFF);

  // pack weights to bf16 B-fragment layout (mw0: first 256 rows; row 256 applied in f32)
  egcl_pack<<<(8*4096+255)/256,256,0,stream>>>(mw0, wp+MW0P, 256, 8);
  egcl_pack<<<(4*4096+255)/256,256,0,stream>>>(mw1, wp+MW1P, 128, 4);
  egcl_pack<<<(4*4096+255)/256,256,0,stream>>>(cw0, wp+CW0P, 128, 4);
  egcl_pack<<<(4*4096+255)/256,256,0,stream>>>(cw1, wp+CW1P, 128, 4);
  egcl_pack<<<(8*4096+255)/256,256,0,stream>>>(nw0, wp+NW0P, 256, 8);
  egcl_pack<<<(4*4096+255)/256,256,0,stream>>>(nw1, wp+NW1P, 128, 4);
  egcl_pack<<<(4*4096+255)/256,256,0,stream>>>(nw2, wp+NW2P, 128, 4);

  hipMemsetAsync(agg,  0, (size_t)Nn*128*4, stream);
  hipMemsetAsync(tsum, 0, (size_t)Nn*3*4,   stream);
  hipMemsetAsync(cnt,  0, (size_t)Nn*4,     stream);

  egcl_edge<<<Ee/64, 128, 0, stream>>>(h, crd, ei, wp, mw0, mb0, mb1, cb0, cb1, cw2, agg, tsum, cnt);
  egcl_node<<<(Nn+63)/64, 128, 0, stream>>>(h, wp, nb0, nb1, nb2, agg, cnt, out);
  egcl_coord<<<(Nn*3+255)/256, 256, 0, stream>>>(crd, tsum, cnt, out + (size_t)Nn*128);
}
